// Encoder_4389456577186
// MI455X (gfx1250) — compile-verified, matrix operand order checked by reading, not run on hardware
//
#include <hip/hip_runtime.h>

// LSTM encoder for MI455X (gfx1250, wave32, WMMA + TDM).
//   1. prep:   W_ih -> bf16; W_hh -> bf16 block-permuted (contiguous 256-row
//              tiles per step-kernel block, TDM-friendly); zero h0/c0.
//   2. embed:  gather embedding rows -> Xb (S,B,H) bf16.
//   3. gemm:   Gx = Xb @ W_ih^T + (b_ih+b_hh), v_wmma_f32_16x16x32_bf16.
//   4. 512x step: gates = h_{t-1} @ W_hh^T + Gx[t]. W_hh tiles stream into
//              LDS via tensor_load_to_lds (double-buffered, TENSORcnt-
//              pipelined); B-fragments read with ds_load_b128; elementwise
//              i,f,g,o / c / h update in registers; h double-buffered.

#define DI __device__ __forceinline__

typedef __attribute__((ext_vector_type(16))) __bf16 v16bf;
typedef __attribute__((ext_vector_type(8)))  __bf16 v8bf;
typedef __attribute__((ext_vector_type(8)))  float  v8f;
typedef unsigned int u32x4 __attribute__((ext_vector_type(4)));
typedef int          i32x4 __attribute__((ext_vector_type(4)));
typedef int          i32x8 __attribute__((ext_vector_type(8)));

constexpr int B  = 64;
constexpr int S  = 512;
constexpr int H  = 512;
constexpr int G4 = 2048;          // 4*H, gate order i,f,g,o

// LDS staging: 256 rows x 32 K-elems (64B) per chunk, padded to 80B rows.
// 80B = 20 dwords: 20*n mod 64 distinct for n=0..15 -> conflict-free ds_load_b128.
constexpr int LDS_ROW_B = 80;
constexpr int LDS_BUF_B = 256 * LDS_ROW_B;   // 20KB per buffer, x2 = 40KB

#if defined(__AMDGCN__) && __has_builtin(__builtin_amdgcn_tensor_load_to_lds) && \
    __has_builtin(__builtin_amdgcn_s_wait_tensorcnt)
#define USE_TDM 1
#else
#define USE_TDM 0
#endif

DI unsigned short f2bf(float f) {
    unsigned u = __float_as_uint(f);
    u += 0x7FFFu + ((u >> 16) & 1u);   // round-to-nearest-even
    return (unsigned short)(u >> 16);
}
DI float bf2f(unsigned short s) { return __uint_as_float(((unsigned)s) << 16); }

DI float sigmoid_(float x) { return 1.0f / (1.0f + __expf(-x)); }
DI float tanh_(float x)    { float e = __expf(2.0f * x); return 1.0f - 2.0f / (e + 1.0f); }

// ISA 7.12.2 16-bit A/B fragment: lane (m = l&15, half = l>>4) holds two
// contiguous 16B K-chunks at elem offsets {half*8} and {16+half*8}.
DI v16bf load_frag(const unsigned short* row, int k0, int half) {
    v8bf lo = *(const v8bf*)(row + k0 + half * 8);
    v8bf hi = *(const v8bf*)(row + k0 + 16 + half * 8);
    return __builtin_shufflevector(lo, hi, 0,1,2,3,4,5,6,7,8,9,10,11,12,13,14,15);
}

// B fragment from a padded LDS chunk buffer: row = n_local, bytes {half*16, 32+half*16}.
DI v16bf lds_frag(const unsigned char* buf, int nrow, int half) {
    const unsigned char* p = buf + nrow * LDS_ROW_B + half * 16;
    v8bf lo = *(const v8bf*)(p);
    v8bf hi = *(const v8bf*)(p + 32);
    return __builtin_shufflevector(lo, hi, 0,1,2,3,4,5,6,7,8,9,10,11,12,13,14,15);
}

#if USE_TDM
// Issue one TDM load: 2D tile (32 K-elems x 256 rows, bf16) from the permuted
// W_hh slice into LDS at lds_byte, with 16B padding after each 64B row.
DI void tdm_issue(const unsigned short* gsrc, unsigned lds_byte) {
    unsigned long long ga = (unsigned long long)gsrc;
    u32x4 g0 = {0u, 0u, 0u, 0u};
    g0[0] = 1u;                                   // count=1, user mode
    g0[1] = lds_byte;                             // lds_addr
    g0[2] = (unsigned)ga;                         // global_addr[31:0]
    g0[3] = (unsigned)(ga >> 32) | 0x80000000u;   // global_addr[56:32] | type=2
    i32x8 g1 = {0, 0, 0, 0, 0, 0, 0, 0};
    // data_size=2B | pad_enable | pad_interval=16dw | pad_amount=4dw
    g1[0] = (1 << 16) | (1 << 20) | (3 << 22) | (3 << 25);
    g1[1] = (int)(512u << 16);                    // tensor_dim0 = 512 (K)
    g1[2] = (int)(256u << 16);                    // tensor_dim1 = 256 rows
    g1[3] = (int)(32u << 16);                     // tile_dim0 = 32 K-elems
    g1[4] = 256;                                  // tile_dim1 = 256 rows
    g1[5] = 512;                                  // tensor_dim0_stride = 512 elems
    i32x4 z4 = {0, 0, 0, 0};
#if __has_include(<hip/amd_detail/amd_gfx1250_TDM.h>)
    i32x8 z8 = {0, 0, 0, 0, 0, 0, 0, 0};
    __builtin_amdgcn_tensor_load_to_lds(g0, g1, z4, z4, z8, 0);   // 6-arg toolchain
#else
    __builtin_amdgcn_tensor_load_to_lds(g0, g1, z4, z4, 0);       // 5-arg (ROCm 7.2)
#endif
}
#endif

// Fallback staging: cooperative load + ds_store into the same padded layout.
DI void stage_fallback(const unsigned short* Wp_blk, int k0, unsigned char* buf, int tid) {
#pragma unroll
    for (int it = 0; it < 8; ++it) {
        int e = it * 128 + tid;                   // 1024 x 16B chunks
        int row = e >> 2, kq = e & 3;
        v8bf v = *(const v8bf*)(Wp_blk + row * H + k0 + kq * 8);
        *(v8bf*)(buf + row * LDS_ROW_B + kq * 16) = v;
    }
}

// ---------------- prep: weights -> bf16 (+ block-permute W_hh), zero h/c ----------------
// Wp layout: block nb (64 hidden units) owns 256 contiguous rows ordered q*64+u,
// i.e. Wp[(nb*256 + q*64 + u)*H + k] = W_hh[(q*512 + nb*64 + u)*H + k].
__global__ void prep_kernel(const float* __restrict__ Wih, const float* __restrict__ Whh,
                            unsigned short* __restrict__ Wihb, unsigned short* __restrict__ Wp,
                            unsigned short* __restrict__ hbuf, float* __restrict__ cbuf) {
    int i = blockIdx.x * blockDim.x + threadIdx.x;   // 0 .. G4*H-1
    Wihb[i] = f2bf(Wih[i]);
    int n = i >> 9, k = i & 511;
    int q = n >> 9, rem = n & 511, nb = rem >> 6, u = rem & 63;
    Wp[((size_t)(nb * 256 + q * 64 + u)) * H + k] = f2bf(Whh[i]);
    if (i < B * H) {
        hbuf[i]         = 0;
        hbuf[B * H + i] = 0;
        cbuf[i]         = 0.0f;
    }
}

// ---------------- embedding gather -> Xb (S,B,H) bf16 ----------------
__global__ void embed_kernel(const int* __restrict__ inp, const float* __restrict__ emb,
                             unsigned short* __restrict__ Xb) {
    int row = blockIdx.x;            // row = t*B + b
    int t = row / B, b = row % B;
    int tok = inp[b * S + t];        // inputs is (B,S)
    const float* src = emb + (long long)tok * H;
    unsigned short* dst = Xb + (long long)row * H;
    for (int k = threadIdx.x; k < H; k += blockDim.x) dst[k] = f2bf(src[k]);
}

// ---------------- Gx = Xb @ W_ih^T + bias (bf16 out) ----------------
__global__ void __launch_bounds__(128) gemm_xw_kernel(
        const unsigned short* __restrict__ Xb, const unsigned short* __restrict__ Wihb,
        const float* __restrict__ bih, const float* __restrict__ bhh,
        unsigned short* __restrict__ Gx) {
    const int lane = threadIdx.x & 31;
    const int w    = threadIdx.x >> 5;
    const int half = lane >> 4;
    const int ln   = lane & 15;
    const int mBase = blockIdx.y * 64 + w * 16;
    const int nBase = blockIdx.x * 64;

    v8f acc[4] = {};
    const unsigned short* arow = Xb + (long long)(mBase + ln) * H;
    for (int k0 = 0; k0 < H; k0 += 32) {
        v16bf a = load_frag(arow, k0, half);
#pragma unroll
        for (int s = 0; s < 4; ++s) {
            const unsigned short* brow = Wihb + (long long)(nBase + s * 16 + ln) * H;
            v16bf bf = load_frag(brow, k0, half);
            acc[s] = __builtin_amdgcn_wmma_f32_16x16x32_bf16(
                false, a, false, bf, (short)0, acc[s], false, false);
        }
    }
#pragma unroll
    for (int s = 0; s < 4; ++s) {
        int n = nBase + s * 16 + ln;
        float bias = bih[n] + bhh[n];
#pragma unroll
        for (int r = 0; r < 8; ++r) {
            int m = mBase + r + 8 * half;          // D layout: M = r + 8*half, N = ln
            Gx[(long long)m * G4 + n] = f2bf(acc[s][r] + bias);
        }
    }
}

// ---------------- one recurrent step ----------------
// grid = H/64 blocks; block owns 64 hidden units x all 4 gates x all batch.
// W_hh slice streams through LDS (double-buffered TDM); A (h) from global.
__global__ void __launch_bounds__(128) lstm_step_kernel(
        const unsigned short* __restrict__ hin, unsigned short* __restrict__ hout,
        float* __restrict__ cbuf, const unsigned short* __restrict__ Wp,
        const unsigned short* __restrict__ Gx, float* __restrict__ out,
        int t, int isLast) {
    __shared__ __align__(16) unsigned char smem[2 * LDS_BUF_B];
    const int tid  = threadIdx.x;
    const int lane = tid & 31;
    const int w    = tid >> 5;
    const int half = lane >> 4;
    const int ln   = lane & 15;
    const int mBase = w * 16;              // batch rows
    const int nBase = blockIdx.x * 64;     // hidden units

    const unsigned short* Wp_blk = Wp + (size_t)blockIdx.x * 256 * H;

#if USE_TDM
    // ISA 10.2: low 32 bits of a generic LDS pointer = wave-relative LDS address.
    const unsigned lds_base = (unsigned)(unsigned long long)&smem[0];
    if (w == 0) tdm_issue(Wp_blk, lds_base);                 // prefetch chunk 0
#else
    stage_fallback(Wp_blk, 0, smem, tid);
#endif

    v8f acc[4][4] = {};
    const unsigned short* arow = hin + (long long)(mBase + ln) * H;

    for (int kc = 0; kc < 16; ++kc) {
        const int k0 = kc * 32;
#if USE_TDM
        if (w == 0) {
            if (kc + 1 < 16) {
                tdm_issue(Wp_blk + (kc + 1) * 32, lds_base + ((kc + 1) & 1) * LDS_BUF_B);
                __builtin_amdgcn_s_wait_tensorcnt(1);        // chunk kc landed
            } else {
                __builtin_amdgcn_s_wait_tensorcnt(0);
            }
        }
        __syncthreads();                                     // publish chunk kc
#else
        if (kc + 1 < 16)
            stage_fallback(Wp_blk, (kc + 1) * 32, smem + ((kc + 1) & 1) * LDS_BUF_B, tid);
        __syncthreads();
#endif
        const unsigned char* buf = smem + (kc & 1) * LDS_BUF_B;
        v16bf a = load_frag(arow, k0, half);                 // A: this wave's h rows
#pragma unroll
        for (int q = 0; q < 4; ++q) {                        // gate i,f,g,o
#pragma unroll
            for (int s = 0; s < 4; ++s) {                    // 16-col sub-tile
                v16bf bf = lds_frag(buf, q * 64 + s * 16 + ln, half);
                acc[q][s] = __builtin_amdgcn_wmma_f32_16x16x32_bf16(
                    false, a, false, bf, (short)0, acc[q][s], false, false);
            }
        }
        __syncthreads();                                     // done reading buf(kc)
    }

    const unsigned short* gxt = Gx + (long long)t * B * G4;
#pragma unroll
    for (int s = 0; s < 4; ++s) {
        int j = nBase + s * 16 + ln;
#pragma unroll
        for (int r = 0; r < 8; ++r) {
            int m = mBase + r + 8 * half;
            const unsigned short* g = gxt + (long long)m * G4;
            float gi = acc[0][s][r] + bf2f(g[0 * H + j]);
            float gf = acc[1][s][r] + bf2f(g[1 * H + j]);
            float gg = acc[2][s][r] + bf2f(g[2 * H + j]);
            float go = acc[3][s][r] + bf2f(g[3 * H + j]);
            float iv = sigmoid_(gi), fv = sigmoid_(gf);
            float gv = tanh_(gg),    ov = sigmoid_(go);
            float c_old = cbuf[m * H + j];
            float c_new = fv * c_old + iv * gv;
            float hv = ov * tanh_(c_new);
            cbuf[m * H + j] = c_new;
            hout[m * H + j] = f2bf(hv);
            out[((long long)m * S + t) * H + j] = hv;        // annotations (B,S,H)
            if (isLast) {
                out[(long long)B * S * H + m * H + j] = hv;                       // h
                out[(long long)B * S * H + (long long)B * H + m * H + j] = c_new; // c
            }
        }
    }
}

extern "C" void kernel_launch(void* const* d_in, const int* in_sizes, int n_in,
                              void* d_out, int out_size, void* d_ws, size_t ws_size,
                              hipStream_t stream) {
    const int*   inp = (const int*)d_in[0];    // (B,S) token ids
    const float* emb = (const float*)d_in[1];  // (VOCAB,H)
    const float* Wih = (const float*)d_in[2];  // (4H,H)
    const float* Whh = (const float*)d_in[3];  // (4H,H)
    const float* bih = (const float*)d_in[4];  // (4H)
    const float* bhh = (const float*)d_in[5];  // (4H)
    float* out = (float*)d_out;

    // workspace layout (16B-aligned): Wihb | Wp | Xb | Gx | hbuf(x2) | c
    unsigned short* Wihb = (unsigned short*)d_ws;
    unsigned short* Wp   = Wihb + (size_t)G4 * H;
    unsigned short* Xb   = Wp   + (size_t)G4 * H;
    unsigned short* Gx   = Xb   + (size_t)S * B * H;
    unsigned short* hbuf = Gx   + (size_t)S * B * G4;
    float*          cbuf = (float*)(hbuf + 2 * (size_t)B * H);

    prep_kernel<<<(G4 * H) / 256, 256, 0, stream>>>(Wih, Whh, Wihb, Wp, hbuf, cbuf);
    embed_kernel<<<S * B, 256, 0, stream>>>(inp, emb, Xb);
    gemm_xw_kernel<<<dim3(G4 / 64, (S * B) / 64), 128, 0, stream>>>(Xb, Wihb, bih, bhh, Gx);

    for (int t = 0; t < S; ++t) {
        const unsigned short* hin = hbuf + (size_t)(t & 1) * B * H;
        unsigned short*      hout = hbuf + (size_t)((t + 1) & 1) * B * H;
        lstm_step_kernel<<<H / 64, 128, 0, stream>>>(
            hin, hout, cbuf, Wp, Gx, out, t, (t == S - 1) ? 1 : 0);
    }
}
